// _GRULayer_72980084294194
// MI455X (gfx1250) — matrix operand order repaired and checked by reading, not compiled, not run on hardware
//
#include <hip/hip_runtime.h>
#include <hip/hip_bf16.h>
#include <stddef.h>

// ---------------------------------------------------------------------------
// GRU layer for MI455X (gfx1250, wave32, WMMA).
//   B=8, T=2048, H=1024, gates=3H=3072.
// Phase 1: x_proj GEMM [16384,1024]x[1024,3072], 32x64 tile per wave,
//          ping-pong (copy-free) double-buffered v_wmma_f32_16x16x32_bf16.
// Phase 2: persistent cooperative kernel, one grid barrier per timestep.
//          Each wave's W_hh slice (8 KB) lives in VGPRs across the t-loop.
// ---------------------------------------------------------------------------

typedef __attribute__((ext_vector_type(16))) __bf16 v16bf;
typedef __attribute__((ext_vector_type(8)))  __bf16 v8bf;
typedef __attribute__((ext_vector_type(4)))  __bf16 v4bf;
typedef __attribute__((ext_vector_type(8)))  float  v8f;
typedef __attribute__((ext_vector_type(4)))  float  v4f;

#define HID    1024
#define G3     3072
#define NBATCH 8
#define TSTEPS 2048
#define BT     (NBATCH * TSTEPS)   // 16384 rows of X

// ---- WMMA operand loaders (contiguous k = 16*(lane/16)+s, same in A and B, wave32) ----------

// A operand: 16x32 bf16 tile, row m = lane&15, row-major source with stride lda.
__device__ __forceinline__ v16bf load_a_tile(const __bf16* __restrict__ A,
                                             int lda, int row, int k0, int lane) {
    const int kb = (lane < 16) ? 0 : 16;
    const size_t base = (size_t)row * lda + k0 + kb;
    v8bf lo = *reinterpret_cast<const v8bf*>(A + base);        // K = kb..kb+7
    v8bf hi = *reinterpret_cast<const v8bf*>(A + base + 8);    // K = kb+8..kb+15
    v16bf r;
#pragma unroll
    for (int i = 0; i < 8; ++i) { r[i] = lo[i]; r[i + 8] = hi[i]; }
    return r;
}

// B operand: 32x16 bf16 tile (KxN). Source W stored [N,K] row-major (ldk=HID),
// so column n of B == row n of W: 16 contiguous K values per lane.
__device__ __forceinline__ v16bf load_b_tile(const __bf16* __restrict__ W,
                                             int col, int k0, int lane) {
    const int kb = (lane < 16) ? 0 : 16;
    return *reinterpret_cast<const v16bf*>(W + (size_t)col * HID + k0 + kb);
}

// ---- fp32 -> bf16 conversion (vectorized) ---------------------------------
__global__ void gru_cvt_bf16(const v4f* __restrict__ in, v4bf* __restrict__ out,
                             size_t n4) {
    size_t i = (size_t)blockIdx.x * blockDim.x + threadIdx.x;
    const size_t stride = (size_t)gridDim.x * blockDim.x;
    for (; i < n4; i += stride) {
        out[i] = __builtin_convertvector(in[i], v4bf);
    }
}

// ---- init hidden state + sync counter -------------------------------------
__global__ void gru_init_state(__bf16* __restrict__ h_bf, float* __restrict__ h_f,
                               unsigned* __restrict__ counter) {
    const int i = blockIdx.x * blockDim.x + threadIdx.x;
    if (i < 16 * HID) h_bf[i] = (__bf16)0.0f;       // padded rows 8..15 stay 0
    if (i < NBATCH * HID) h_f[i] = 0.0f;
    if (i == 0) *counter = 0u;
}

// ---- Phase 1: x_proj = X @ W_ih^T + b_ih ----------------------------------
// Each wave computes a 32(M) x 64(N) f32 tile. K stepped by 64 with two
// alternating operand register sets (ping-pong): zero loop-carried copies.
__global__ void __launch_bounds__(256)
gru_xproj_gemm(const __bf16* __restrict__ Xb,    // [BT, HID] bf16
               const __bf16* __restrict__ Wb,    // [G3, HID] bf16
               const float*  __restrict__ bias,  // [G3]
               float* __restrict__ out) {        // [BT, G3] f32
    const int lane = threadIdx.x & 31;
    const int wave = (int)((blockIdx.x * blockDim.x + threadIdx.x) >> 5);
    const int NCH  = G3 / 64;                    // 48 N-chunks
    const int mch  = wave / NCH;                 // 512 M-chunks of 32 rows
    const int nch  = wave % NCH;
    if (mch >= BT / 32) return;

    const int row0  = mch * 32 + (lane & 15);
    const int row1  = row0 + 16;
    const int nbase = nch * 64;
    const int ncol  = lane & 15;

    v8f acc[2][4];
#pragma unroll
    for (int mt = 0; mt < 2; ++mt)
#pragma unroll
        for (int nt = 0; nt < 4; ++nt) acc[mt][nt] = v8f{};

    v16bf aA[2], bA[4], aB[2], bB[4];

    // prologue: set A holds K-slice 0
    aA[0] = load_a_tile(Xb, HID, row0, 0, lane);
    aA[1] = load_a_tile(Xb, HID, row1, 0, lane);
#pragma unroll
    for (int nt = 0; nt < 4; ++nt)
        bA[nt] = load_b_tile(Wb, nbase + nt * 16 + ncol, 0, lane);

    for (int k0 = 0; k0 < HID; k0 += 64) {
        // issue loads for slice k0+32 into set B, then compute with set A
        const int k1 = (k0 + 32 < HID) ? (k0 + 32) : 0;
        aB[0] = load_a_tile(Xb, HID, row0, k1, lane);
        aB[1] = load_a_tile(Xb, HID, row1, k1, lane);
#pragma unroll
        for (int nt = 0; nt < 4; ++nt)
            bB[nt] = load_b_tile(Wb, nbase + nt * 16 + ncol, k1, lane);
#pragma unroll
        for (int mt = 0; mt < 2; ++mt)
#pragma unroll
            for (int nt = 0; nt < 4; ++nt)
                acc[mt][nt] = __builtin_amdgcn_wmma_f32_16x16x32_bf16(
                    false, aA[mt], false, bA[nt], (short)0, acc[mt][nt],
                    false, false);

        // issue loads for slice k0+64 into set A, then compute with set B
        const int k2 = (k0 + 64 < HID) ? (k0 + 64) : 0;
        aA[0] = load_a_tile(Xb, HID, row0, k2, lane);
        aA[1] = load_a_tile(Xb, HID, row1, k2, lane);
#pragma unroll
        for (int nt = 0; nt < 4; ++nt)
            bA[nt] = load_b_tile(Wb, nbase + nt * 16 + ncol, k2, lane);
#pragma unroll
        for (int mt = 0; mt < 2; ++mt)
#pragma unroll
            for (int nt = 0; nt < 4; ++nt)
                acc[mt][nt] = __builtin_amdgcn_wmma_f32_16x16x32_bf16(
                    false, aB[mt], false, bB[nt], (short)0, acc[mt][nt],
                    false, false);
    }

#pragma unroll
    for (int nt = 0; nt < 4; ++nt) {
        const int n  = nbase + nt * 16 + ncol;
        const float bv = bias[n];
#pragma unroll
        for (int mt = 0; mt < 2; ++mt) {
            const int rbase = mch * 32 + mt * 16;
#pragma unroll
            for (int v = 0; v < 8; ++v) {
                const int m = (lane < 16) ? v : (v + 8);   // C/D layout
                out[(size_t)(rbase + m) * G3 + n] = acc[mt][nt][v] + bv;
            }
        }
    }
}

// ---- Phase 2: persistent recurrent scan -----------------------------------
// 64 blocks x 384 threads (12 waves). Block owns hidden units
// [16*blk, 16*blk+16). Wave w: gate = w%3 (r/z/n), K-quarter q = w/3.
// W_hh slice (8 iters x 8 VGPRs = 64 VGPRs) preloaded before the t-loop.
__global__ void __launch_bounds__(384, 1)
gru_recurrent(const __bf16* __restrict__ Whh_b,  // [G3, HID] bf16
              const float*  __restrict__ b_hh,   // [G3]
              const float*  __restrict__ xproj,  // [NBATCH, TSTEPS, G3] f32
              __bf16* __restrict__ h_bf,         // [16, HID] (rows 8..15 == 0)
              float*  __restrict__ h_f,          // [NBATCH, HID]
              float*  __restrict__ out,          // [NBATCH, TSTEPS, HID]
              unsigned* __restrict__ counter) {
    __shared__ float part[12][NBATCH][16];

    const int tid   = threadIdx.x;
    const int lane  = tid & 31;
    const int w     = tid >> 5;          // 0..11
    const int gate  = w % 3;
    const int q     = w / 3;             // K quarter
    const int jbase = blockIdx.x * 16;
    const int col   = gate * HID + jbase + (lane & 15);
    const int arow  = lane & 15;         // batch row (8..15 are zero padding)
    const unsigned nblocks = gridDim.x;
    const int kstart = q * 256;

    // Hoist this wave's W_hh slice into registers: invariant across all steps.
    v16bf breg[8];
#pragma unroll
    for (int i = 0; i < 8; ++i)
        breg[i] = load_b_tile(Whh_b, col, kstart + i * 32, lane);

    // Hoist biases for the elementwise threads (invariant across steps).
    float bias_g[3] = {0.f, 0.f, 0.f};
    if (tid < 128) {
        const int jj = tid & 15;
#pragma unroll
        for (int g = 0; g < 3; ++g) bias_g[g] = b_hh[g * HID + jbase + jj];
    }

    for (int t = 0; t < TSTEPS; ++t) {
        // g_h partial = h @ W_hh^T over this wave's K quarter.
        // All 8 A-tile loads are independent -> issued up front, waits overlap.
        v8f acc = {};
#pragma unroll
        for (int i = 0; i < 8; ++i) {
            v16bf a = load_a_tile(h_bf, HID, arow, kstart + i * 32, lane);
            acc = __builtin_amdgcn_wmma_f32_16x16x32_bf16(
                false, a, false, breg[i], (short)0, acc, false, false);
        }
        if (lane < 16) {                  // lanes 0..15 hold batches 0..7
#pragma unroll
            for (int v = 0; v < NBATCH; ++v) part[w][v][lane] = acc[v];
        }
        __syncthreads();

        if (tid < 128) {                  // one thread per (batch, hidden unit)
            const int b  = tid >> 4;
            const int jj = tid & 15;
            float gh[3];
#pragma unroll
            for (int g = 0; g < 3; ++g) {
                float s = bias_g[g];
#pragma unroll
                for (int qq = 0; qq < 4; ++qq) s += part[qq * 3 + g][b][jj];
                gh[g] = s;
            }
            const size_t xb = ((size_t)b * TSTEPS + t) * G3 + jbase + jj;
            const float gx_r = xproj[xb];
            const float gx_z = xproj[xb + HID];
            const float gx_n = xproj[xb + 2 * HID];
            if (t + 1 < TSTEPS) {          // pull next step's gates toward L2/L0
                __builtin_prefetch(&xproj[xb + G3], 0, 1);
                __builtin_prefetch(&xproj[xb + G3 + HID], 0, 1);
                __builtin_prefetch(&xproj[xb + G3 + 2 * HID], 0, 1);
            }

            const float r  = 1.0f / (1.0f + __expf(-(gx_r + gh[0])));
            const float z  = 1.0f / (1.0f + __expf(-(gx_z + gh[1])));
            const float nn = tanhf(gx_n + r * gh[2]);
            const float ho = h_f[b * HID + jbase + jj];
            const float hn = (1.0f - z) * nn + z * ho;

            h_f[b * HID + jbase + jj]  = hn;
            h_bf[b * HID + jbase + jj] = (__bf16)hn;
            out[((size_t)b * TSTEPS + t) * HID + jbase + jj] = hn;
        }
        __syncthreads();

        // grid-wide barrier: release h writes, arrive, spin, acquire
        if (tid == 0) {
            __threadfence();
            atomicAdd(counter, 1u);
            const unsigned target = nblocks * (unsigned)(t + 1);
            while (__hip_atomic_load(counter, __ATOMIC_RELAXED,
                                     __HIP_MEMORY_SCOPE_AGENT) < target) {
                __builtin_amdgcn_s_sleep(2);
            }
        }
        __syncthreads();
        __threadfence();                  // invalidate stale h in near caches
    }
}

// ---------------------------------------------------------------------------
extern "C" void kernel_launch(void* const* d_in, const int* in_sizes, int n_in,
                              void* d_out, int out_size, void* d_ws, size_t ws_size,
                              hipStream_t stream) {
    (void)in_sizes; (void)n_in; (void)out_size; (void)ws_size;
    const float* x    = (const float*)d_in[0];   // [8,2048,1024]
    const float* W_ih = (const float*)d_in[1];   // [3072,1024]
    const float* b_ih = (const float*)d_in[2];   // [3072]
    const float* W_hh = (const float*)d_in[3];   // [3072,1024]
    const float* b_hh = (const float*)d_in[4];   // [3072]
    float* out = (float*)d_out;                  // [8,2048,1024]

    // Workspace layout (bytes), all offsets 4KB-aligned:
    char* ws = (char*)d_ws;
    __bf16* Xb    = (__bf16*)(ws);                    //  33,554,432  X bf16
    __bf16* Wihb  = (__bf16*)(ws +  33554432);        //   6,291,456  W_ih bf16
    __bf16* Whhb  = (__bf16*)(ws +  39845888);        //   6,291,456  W_hh bf16
    float*  xproj = (float* )(ws +  46137344);        // 201,326,592  x_proj f32
    __bf16* h_bf  = (__bf16*)(ws + 247463936);        //      32,768  h (16x1024)
    float*  h_f   = (float* )(ws + 247496704);        //      32,768  h f32
    unsigned* cnt = (unsigned*)(ws + 247529472);      //           4  barrier

    // fp32 -> bf16 conversions
    gru_cvt_bf16<<<2048, 256, 0, stream>>>((const v4f*)x,    (v4bf*)Xb,
                                           (size_t)BT * HID / 4);
    gru_cvt_bf16<<<256, 256, 0, stream>>>((const v4f*)W_ih, (v4bf*)Wihb,
                                          (size_t)G3 * HID / 4);
    gru_cvt_bf16<<<256, 256, 0, stream>>>((const v4f*)W_hh, (v4bf*)Whhb,
                                          (size_t)G3 * HID / 4);
    gru_init_state<<<64, 256, 0, stream>>>(h_bf, h_f, cnt);

    // Phase 1: 512 M-chunks x 48 N-chunks = 24576 waves = 3072 blocks x 8 waves
    gru_xproj_gemm<<<3072, 256, 0, stream>>>(Xb, Wihb, b_ih, xproj);

    // Phase 2: persistent scan, 64 blocks (one per 16 hidden units) x 12 waves
    gru_recurrent<<<64, 384, 0, stream>>>(Whhb, b_hh, xproj, h_bf, h_f, out, cnt);
}
